// MultiHeadSelfAttention_41205916237932
// MI455X (gfx1250) — matrix operand order repaired and checked by reading, not compiled, not run on hardware
//
#include <hip/hip_runtime.h>
#include <hip/hip_bf16.h>

typedef __bf16 bf16;
typedef bf16 bf16x16 __attribute__((ext_vector_type(16)));
typedef bf16 bf16x8  __attribute__((ext_vector_type(8)));
typedef float f32x8  __attribute__((ext_vector_type(8)));
typedef int   i32x4  __attribute__((vector_size(16)));   // matches builtin proto

static constexpr int B = 4, S = 1024, D = 1024, H = 16, DH = 64;

__device__ __forceinline__ f32x8 wmma_bf16(bf16x16 a, bf16x16 b, f32x8 c) {
    return __builtin_amdgcn_wmma_f32_16x16x32_bf16(false, a, false, b, (short)0, c,
                                                   false, false);
}

__device__ __forceinline__ bf16x16 cat(bf16x8 lo, bf16x8 hi) {
    return __builtin_shufflevector(lo, hi, 0,1,2,3,4,5,6,7,8,9,10,11,12,13,14,15);
}

// convert 8 contiguous f32 -> bf16x8
__device__ __forceinline__ bf16x8 cvt8(const float* __restrict__ p) {
    float4 a = *(const float4*)p;
    float4 b = *(const float4*)(p + 4);
    bf16x8 r;
    r[0]=(bf16)a.x; r[1]=(bf16)a.y; r[2]=(bf16)a.z; r[3]=(bf16)a.w;
    r[4]=(bf16)b.x; r[5]=(bf16)b.y; r[6]=(bf16)b.z; r[7]=(bf16)b.w;
    return r;
}

__device__ __forceinline__ f32x8 zero8() {
    f32x8 z;
    #pragma unroll
    for (int v = 0; v < 8; ++v) z[v] = 0.0f;
    return z;
}

// ---- CDNA5 async global->LDS staging (ASYNCcnt path), with sync fallback ----
#if __has_builtin(__builtin_amdgcn_global_load_async_to_lds_b128)
#define HAVE_ASYNC_LDS 1
#else
#define HAVE_ASYNC_LDS 0
#endif

__device__ __forceinline__ void async_copy16(const bf16* gsrc, bf16* lds_dst) {
#if HAVE_ASYNC_LDS
    __builtin_amdgcn_global_load_async_to_lds_b128(
        (__attribute__((address_space(1))) i32x4*)gsrc,
        (__attribute__((address_space(3))) i32x4*)lds_dst,
        0, 0);
#else
    *(float4*)lds_dst = *(const float4*)gsrc;   // ds_store_b128 fallback
#endif
}

#if __has_builtin(__builtin_amdgcn_s_wait_asynccnt)
#define WAIT_ASYNC(n) __builtin_amdgcn_s_wait_asynccnt(n)
#else
#define WAIT_ASYNC(n) asm volatile("s_wait_asynccnt %0" :: "n"(n) : "memory")
#endif

// ---------------------------------------------------------------------------
// Kernel 1: per-head QKV projection, fp32 in -> bf16 out.
// One wave handles one (b, h, 16-row tile). Q,K stored [BH][S][64];
// V stored transposed [BH][64][S] so the PV B-operand is contiguous later.
// ---------------------------------------------------------------------------
__global__ __launch_bounds__(256) void qkv_kernel(
    const float* __restrict__ x,
    const float* __restrict__ Wq, const float* __restrict__ Wk,
    const float* __restrict__ Wv,
    const float* __restrict__ bq, const float* __restrict__ bk,
    const float* __restrict__ bv,
    bf16* __restrict__ qb, bf16* __restrict__ kbuf, bf16* __restrict__ vtb)
{
    const int lane = threadIdx.x & 31;
    const int w    = (blockIdx.x * blockDim.x + threadIdx.x) >> 5; // 0..4095
    const int mt   = w & (S / 16 - 1);
    const int h    = (w >> 6) & (H - 1);
    const int b    = w >> 10;
    const int half = lane >> 4;
    const int l16  = lane & 15;
    const int s0   = mt * 16;
    const int kb8  = half * 8;
    const int bh   = b * H + h;

    // A-operands: X rows, two K-tiles of 32 (16-bit A 16x32 layout)
    const float* xr = x + ((size_t)(b * S + s0 + l16)) * D + h * DH;
    bf16x16 a0 = cat(cvt8(xr + half * 16),      cvt8(xr + half * 16 + 8));
    bf16x16 a1 = cat(cvt8(xr + 32 + half * 16), cvt8(xr + 32 + half * 16 + 8));

    const float* Ws[3] = { Wq + h * DH * DH, Wk + h * DH * DH, Wv + h * DH * DH };
    const float* bs[3] = { bq + h * DH,      bk + h * DH,      bv + h * DH };

    #pragma unroll
    for (int mtx = 0; mtx < 3; ++mtx) {
        #pragma unroll
        for (int e = 0; e < 4; ++e) {            // output-feature tiles of 16
            const float bias = bs[mtx][e * 16 + l16];
            f32x8 acc;
            #pragma unroll
            for (int v = 0; v < 8; ++v) acc[v] = bias;   // D[m][n] += b[n]
            #pragma unroll
            for (int t = 0; t < 2; ++t) {        // K-tiles over d_head
                // B[k][n] = W[n][k]: 16 contiguous floats of row (e*16+l16)
                const float* wr = Ws[mtx] + (e * 16 + l16) * DH + t * 32 + half * 16;
                bf16x16 bmat = cat(cvt8(wr), cvt8(wr + 8));
                acc = wmma_bf16(t ? a1 : a0, bmat, acc);
            }
            if (mtx == 2) {                      // V^T store: [BH][64][S]
                #pragma unroll
                for (int v = 0; v < 8; ++v)
                    vtb[((size_t)bh * DH + e * 16 + l16) * S + s0 + v + 8 * half] =
                        (bf16)acc[v];
            } else {                             // Q / K store: [BH][S][64]
                bf16* dst = (mtx == 0) ? qb : kbuf;
                #pragma unroll
                for (int v = 0; v < 8; ++v)
                    dst[((size_t)bh * S + s0 + v + 8 * half) * DH + e * 16 + l16] =
                        (bf16)acc[v];
            }
        }
    }
}

// ---------------------------------------------------------------------------
// Kernel 2: flash attention. 8 waves/block share one (b,h); K/V tiles are
// staged once per block into LDS with async global->LDS copies (double
// buffered, ASYNCcnt-pipelined). Each wave owns a 16-query tile:
// QK^T (4 wmma) -> online softmax -> P via per-wave LDS swap -> PV (4 wmma).
// ---------------------------------------------------------------------------
__global__ __launch_bounds__(256) void attn_kernel(
    const bf16* __restrict__ qb, const bf16* __restrict__ kbuf,
    const bf16* __restrict__ vtb, float* __restrict__ out)
{
    __shared__ __attribute__((aligned(16))) bf16 k_lds[2][32][64]; // 2 x 4 KB
    __shared__ __attribute__((aligned(16))) bf16 v_lds[2][64][32]; // 2 x 4 KB
    __shared__ __attribute__((aligned(16))) bf16 p_lds[8][16][32]; // 8 x 1 KB

    const int tid  = threadIdx.x;
    const int lane = tid & 31;
    const int wave = tid >> 5;
    const int w    = (blockIdx.x * blockDim.x + tid) >> 5;  // 0..4095
    const int qt   = w & (S / 16 - 1);
    const int h    = (w >> 6) & (H - 1);
    const int b    = w >> 10;
    const int half = lane >> 4;
    const int l16  = lane & 15;
    const int q0   = qt * 16;
    const int kb8  = half * 8;
    const int bh   = b * H + h;
    const float scale = 0.125f;                  // 1/sqrt(64)

    const bf16* kp = kbuf + (size_t)bh * S * DH; // [S][64], keys contiguous
    const bf16* vp = vtb  + (size_t)bh * DH * S; // [64][S]

    // staging work split: K tile = 4KB contiguous; V tile = 64 rows x 64B
    const int vrow = tid >> 2, vchunk = (tid & 3) * 8;

    auto stage = [&](int tile, int buf) {
        async_copy16(kp + (size_t)tile * 32 * DH + tid * 8, &k_lds[buf][0][0] + tid * 8);
        async_copy16(vp + (size_t)vrow * S + tile * 32 + vchunk, &v_lds[buf][vrow][vchunk]);
    };

    // Q A-operands, resident for the whole key loop
    const bf16* qr = qb + ((size_t)bh * S + q0 + l16) * DH;
    bf16x16 aq0 = cat(*(const bf16x8*)(qr + half * 16),
                      *(const bf16x8*)(qr + half * 16 + 8));
    bf16x16 aq1 = cat(*(const bf16x8*)(qr + 32 + half * 16),
                      *(const bf16x8*)(qr + 32 + half * 16 + 8));

    float m[8], l[8];
    f32x8 o[4];
    #pragma unroll
    for (int v = 0; v < 8; ++v) { m[v] = -INFINITY; l[v] = 0.0f; }
    #pragma unroll
    for (int e = 0; e < 4; ++e) o[e] = zero8();

    stage(0, 0);                                 // prologue fill

    for (int it = 0; it < S / 32; ++it) {        // 32 keys per iteration
        if (it + 1 < S / 32) {
            stage(it + 1, (it + 1) & 1);
            WAIT_ASYNC(2);                       // in-order: tile `it` landed
        } else {
            WAIT_ASYNC(0);
        }
        __syncthreads();
        const int buf = it & 1;

        // --- scores: two 16x16 tiles over d_head=64 (2 K-steps each) ---
        bf16x16 b00 = cat(*(const bf16x8*)&k_lds[buf][l16][half * 16],
                          *(const bf16x8*)&k_lds[buf][l16][half * 16 + 8]);
        bf16x16 b01 = cat(*(const bf16x8*)&k_lds[buf][l16][32 + half * 16],
                          *(const bf16x8*)&k_lds[buf][l16][32 + half * 16 + 8]);
        bf16x16 b10 = cat(*(const bf16x8*)&k_lds[buf][16 + l16][half * 16],
                          *(const bf16x8*)&k_lds[buf][16 + l16][half * 16 + 8]);
        bf16x16 b11 = cat(*(const bf16x8*)&k_lds[buf][16 + l16][32 + half * 16],
                          *(const bf16x8*)&k_lds[buf][16 + l16][32 + half * 16 + 8]);

        f32x8 s0 = wmma_bf16(aq1, b01, wmma_bf16(aq0, b00, zero8()));
        f32x8 s1 = wmma_bf16(aq1, b11, wmma_bf16(aq0, b10, zero8()));

        // --- online softmax (row r = v + 8*half lives across 16 lanes) ---
        float mx[8], p0[8], p1[8], rs[8], alpha[8];
        #pragma unroll
        for (int v = 0; v < 8; ++v) {
            s0[v] *= scale; s1[v] *= scale;
            mx[v] = fmaxf(s0[v], s1[v]);
        }
        #pragma unroll
        for (int sh = 1; sh < 16; sh <<= 1)
            #pragma unroll
            for (int v = 0; v < 8; ++v)
                mx[v] = fmaxf(mx[v], __shfl_xor(mx[v], sh, 32));
        #pragma unroll
        for (int v = 0; v < 8; ++v) {
            float mn = fmaxf(m[v], mx[v]);
            alpha[v] = __expf(m[v] - mn);
            m[v] = mn;
            p0[v] = __expf(s0[v] - mn);
            p1[v] = __expf(s1[v] - mn);
            rs[v] = p0[v] + p1[v];
        }
        #pragma unroll
        for (int sh = 1; sh < 16; sh <<= 1)
            #pragma unroll
            for (int v = 0; v < 8; ++v)
                rs[v] += __shfl_xor(rs[v], sh, 32);
        #pragma unroll
        for (int v = 0; v < 8; ++v) l[v] = l[v] * alpha[v] + rs[v];
        #pragma unroll
        for (int e = 0; e < 4; ++e)
            #pragma unroll
            for (int v = 0; v < 8; ++v) o[e][v] *= alpha[v];

        // --- P: C/D layout -> A layout through per-wave LDS ---
        #pragma unroll
        for (int v = 0; v < 8; ++v) {
            p_lds[wave][v + 8 * half][l16]      = (bf16)p0[v];
            p_lds[wave][v + 8 * half][16 + l16] = (bf16)p1[v];
        }
        asm volatile("s_wait_dscnt 0" ::: "memory");  // same-wave RAW on LDS
        bf16x16 pa = cat(*(const bf16x8*)&p_lds[wave][l16][half * 16],
                         *(const bf16x8*)&p_lds[wave][l16][half * 16 + 8]);

        // --- PV: B[k][n] = V^T[n][k], served from the shared V tile ---
        #pragma unroll
        for (int e = 0; e < 4; ++e) {
            bf16x16 bv = cat(*(const bf16x8*)&v_lds[buf][e * 16 + l16][half * 16],
                             *(const bf16x8*)&v_lds[buf][e * 16 + l16][half * 16 + 8]);
            o[e] = wmma_bf16(pa, bv, o[e]);
        }
        __syncthreads();                          // protect buffer reuse
    }

    // --- epilogue: normalize and store fp32 output [B][S][D] ---
    #pragma unroll
    for (int v = 0; v < 8; ++v) {
        const float inv = 1.0f / l[v];
        const int row = q0 + v + 8 * half;
        float* orow = out + ((size_t)(b * S + row)) * D + h * DH;
        #pragma unroll
        for (int e = 0; e < 4; ++e)
            orow[e * 16 + l16] = o[e][v] * inv;
    }
}

extern "C" void kernel_launch(void* const* d_in, const int* in_sizes, int n_in,
                              void* d_out, int out_size, void* d_ws, size_t ws_size,
                              hipStream_t stream) {
    const float* x  = (const float*)d_in[0];
    const float* Wq = (const float*)d_in[1];
    const float* Wk = (const float*)d_in[2];
    const float* Wv = (const float*)d_in[3];
    const float* bq = (const float*)d_in[4];
    const float* bk = (const float*)d_in[5];
    const float* bv = (const float*)d_in[6];
    float* out = (float*)d_out;

    bf16* ws = (bf16*)d_ws;
    const size_t n = (size_t)B * H * S * DH;     // 4M elems per tensor
    bf16* qbp  = ws;
    bf16* kbp  = ws + n;
    bf16* vtbp = ws + 2 * n;

    const int total_waves = B * H * (S / 16);    // 4096
    dim3 blk(256);                                // 8 waves / block
    dim3 grd(total_waves / 8);                   // 512 blocks

    qkv_kernel<<<grd, blk, 0, stream>>>(x, Wq, Wk, Wv, bq, bk, bv, qbp, kbp, vtbp);
    attn_kernel<<<grd, blk, 0, stream>>>(qbp, kbp, vtbp, out);
}
